// MambaBlock_23613730193629
// MI455X (gfx1250) — matrix operand order repaired and checked by reading, not compiled, not run on hardware
//
#include <hip/hip_runtime.h>
#include <hip/hip_bf16.h>

// ---------- WMMA types (gfx1250, wave32) ----------
typedef __attribute__((ext_vector_type(16))) __bf16 v16bf;
typedef __attribute__((ext_vector_type(8)))  float  v8f;
typedef __attribute__((ext_vector_type(2)))  __bf16 bf16x2;
typedef __attribute__((ext_vector_type(4)))  __bf16 bf16x4;

#define TILE_M 128
#define TILE_N 64
#define TILE_K 32
#define LDK    36   // +4 bf16 pad: kills LDS bank conflicts on fragment gathers

// Row-major C[M,N] = A[M,K] * B[K,N]; f32 in/out, bf16 WMMA compute, f32 accum.
// Block = 256 threads = 8 waves. 128x64 tile = 32 subtiles of 16x16; wave w owns
// col (w&3), rows {(w>>2)+2j, j=0..3}: 4 accumulators, one B-fragment reused by
// 4 v_wmma per K-step. Double-buffered LDS: global loads of panel k+1 issue
// before the WMMAs of panel k; one barrier per K-step.
// Requires M%128==0, K%32==0 (true for all calls). NEDGE: branch-free N edge.
template <bool NEDGE>
__global__ __launch_bounds__(256)
void gemm_bf16_wmma(const float* __restrict__ A, const float* __restrict__ B,
                    float* __restrict__ C, int M, int N, int K)
{
    __shared__ __bf16 As[2][TILE_M][LDK];   // [buf][m][k]
    __shared__ __bf16 Bs[2][TILE_N][LDK];   // [buf][n][k] (transposed)

    const int tid  = threadIdx.x;
    const int lane = tid & 31;
    const int wid  = tid >> 5;
    const int m0   = blockIdx.y * TILE_M;
    const int n0   = blockIdx.x * TILE_N;

    const int subc   = (wid & 3) << 4;     // subtile col offset
    const int rowgrp = wid >> 2;           // 0/1 -> rows {rowgrp+2j}
    const int mrow   = lane & 15;
    const int klo    = (lane >> 4) << 3;   // A-frag K offset per lane half
    const int kb     = (lane >> 4) << 4;   // unused; B-frag also takes klo (same k order as A)

    v8f acc[4] = {v8f{}, v8f{}, v8f{}, v8f{}};

    // staging registers (distinct regs -> loads batch, single wait before stores)
    float4 areg[4];
    float  b0reg[4], b1reg[4];

    auto load_panels = [&](int k0) {
        #pragma unroll
        for (int i = 0; i < 4; ++i) {                 // A: 128x32, 4 float4/thread
            int e = tid + (i << 8);
            int m = e >> 3, q = (e & 7) << 2;
            areg[i] = *(const float4*)(A + (size_t)(m0 + m) * K + k0 + q);
        }
        #pragma unroll
        for (int i = 0; i < 4; ++i) {                 // B: 32x64, k-pairs on col
            int e = tid + (i << 8);
            int n = e & 63, kp = e >> 6;
            int gn = n0 + n;
            int cn = NEDGE ? (gn < N ? gn : N - 1) : gn;
            const float* bp = B + (size_t)(k0 + 2 * kp) * N + cn;
            float v0 = bp[0];
            float v1 = bp[N];
            if (NEDGE && gn >= N) { v0 = 0.0f; v1 = 0.0f; }   // v_cndmask
            b0reg[i] = v0; b1reg[i] = v1;
        }
    };
    auto store_panels = [&](int buf) {
        #pragma unroll
        for (int i = 0; i < 4; ++i) {
            int e = tid + (i << 8);
            int m = e >> 3, q = (e & 7) << 2;
            bf16x4 p;
            p[0] = (__bf16)areg[i].x; p[1] = (__bf16)areg[i].y;
            p[2] = (__bf16)areg[i].z; p[3] = (__bf16)areg[i].w;
            *(bf16x4*)&As[buf][m][q] = p;
        }
        #pragma unroll
        for (int i = 0; i < 4; ++i) {
            int e = tid + (i << 8);
            int n = e & 63, kp = e >> 6;
            bf16x2 p; p[0] = (__bf16)b0reg[i]; p[1] = (__bf16)b1reg[i];
            *(bf16x2*)&Bs[buf][n][2 * kp] = p;
        }
    };

    load_panels(0);
    store_panels(0);
    __syncthreads();

    const int nk = K / TILE_K;
    for (int kt = 0; kt < nk; ++kt) {
        const int buf = kt & 1;

        if (kt + 1 < nk) {
            load_panels((kt + 1) * TILE_K);           // issue global loads early
            if (kt + 2 < nk && tid < TILE_M)          // warm L2 two panels ahead
                __builtin_prefetch(A + (size_t)(m0 + tid) * K + (kt + 2) * TILE_K, 0, 1);
        }

        // ---- gather all 5 fragments first, then drain 4 WMMAs ----
        v16bf bfrag;
        #pragma unroll
        for (int i = 0; i < 16; ++i)
            bfrag[i] = Bs[buf][subc + mrow][klo + ((i >> 3) << 4) + (i & 7)];

        v16bf af0, af1, af2, af3;
        #pragma unroll
        for (int i = 0; i < 16; ++i) {
            int k = klo + ((i >> 3) << 4) + (i & 7);
            af0[i] = As[buf][(rowgrp << 4) + mrow][k];
            af1[i] = As[buf][((rowgrp + 2) << 4) + mrow][k];
            af2[i] = As[buf][((rowgrp + 4) << 4) + mrow][k];
            af3[i] = As[buf][((rowgrp + 6) << 4) + mrow][k];
        }
        acc[0] = __builtin_amdgcn_wmma_f32_16x16x32_bf16(false, af0, false, bfrag,
                                                         (short)0, acc[0], false, false);
        acc[1] = __builtin_amdgcn_wmma_f32_16x16x32_bf16(false, af1, false, bfrag,
                                                         (short)0, acc[1], false, false);
        acc[2] = __builtin_amdgcn_wmma_f32_16x16x32_bf16(false, af2, false, bfrag,
                                                         (short)0, acc[2], false, false);
        acc[3] = __builtin_amdgcn_wmma_f32_16x16x32_bf16(false, af3, false, bfrag,
                                                         (short)0, acc[3], false, false);

        if (kt + 1 < nk) store_panels(buf ^ 1);       // write other buffer
        __syncthreads();                              // single barrier per K-step
    }

    // ---- store: D layout -> VGPR jj, lane l: M = jj + 8*(l>=16), N = l&15 ----
    const int colg = n0 + subc + mrow;
    const int rowb = (lane >> 4) << 3;
    if (!NEDGE || colg < N) {
        #pragma unroll
        for (int j = 0; j < 4; ++j) {
            int gm = m0 + ((rowgrp + 2 * j) << 4) + rowb;
            float* cp = C + (size_t)gm * N + colg;
            #pragma unroll
            for (int jj = 0; jj < 8; ++jj)
                cp[(size_t)jj * N] = acc[j][jj];
        }
    }
}

__device__ __forceinline__ float silu_f(float x) {
    return x * (1.0f / (1.0f + __expf(-x)));
}

// Causal depthwise conv1d (K=4) + bias + SiLU over x_gate = xz[..., :D].
// 4 channels per thread; branch-free via clamped row + mask multiply.
__global__ __launch_bounds__(256)
void conv_silu_kernel(const float* __restrict__ xz, const float* __restrict__ conv_w,
                      const float* __restrict__ conv_b, float* __restrict__ xconv,
                      int Ll, int Dd, long total4)
{
    long idx4 = (long)blockIdx.x * 256 + threadIdx.x;
    if (idx4 >= total4) return;
    const int dq = Dd >> 2;
    int d = (int)(idx4 % dq) << 2;
    long bl = idx4 / dq;
    int l = (int)(bl % Ll);
    long brow = bl - l;                       // b*Ll
    const long rs = 2L * Dd;

    float4 acc = *(const float4*)(conv_b + d);
    #pragma unroll
    for (int k = 0; k < 4; ++k) {
        int lt = l - 3 + k;
        float msk = (lt >= 0) ? 1.0f : 0.0f;
        int lc = (lt >= 0) ? lt : 0;
        float4 xv = *(const float4*)(xz + (brow + lc) * rs + d);
        acc.x += msk * conv_w[(d + 0) * 4 + k] * xv.x;
        acc.y += msk * conv_w[(d + 1) * 4 + k] * xv.y;
        acc.z += msk * conv_w[(d + 2) * 4 + k] * xv.z;
        acc.w += msk * conv_w[(d + 3) * 4 + k] * xv.w;
    }
    float4 r;
    r.x = silu_f(acc.x); r.y = silu_f(acc.y);
    r.z = silu_f(acc.z); r.w = silu_f(acc.w);
    *(float4*)(xconv + bl * Dd + d) = r;
}

// Selective scan, fused: softplus(dt), h[16] recurrence in registers, D-skip,
// SiLU(z) gating. One lane per (b,d). No LDS/barriers: B_t/C_t are batch-uniform
// (coalescer broadcast). All t+1 global loads software-pipelined over the
// 16-deep exp chain of step t.
__global__ __launch_bounds__(64)
void scan_kernel(const float* __restrict__ dtlin, const float* __restrict__ b_dt,
                 const float* __restrict__ BC, const float* __restrict__ xconv,
                 const float* __restrict__ A_log, const float* __restrict__ D_param,
                 const float* __restrict__ xz, float* __restrict__ g,
                 int Ll, int Dd)
{
    const int b = blockIdx.x;
    const int d = blockIdx.y * 64 + threadIdx.x;

    float Amat[16];
    #pragma unroll
    for (int s = 0; s < 16; ++s) Amat[s] = -__expf(A_log[d * 16 + s]);
    const float Dp   = D_param[d];
    const float bias = b_dt[d];

    float h[16];
    #pragma unroll
    for (int s = 0; s < 16; ++s) h[s] = 0.0f;

    const long baseBL = (long)b * Ll;
    long off  = baseBL * Dd + d;
    long zoff = baseBL * (2L * Dd) + Dd + d;

    // preload t = 0
    float dt_n = dtlin[off];
    float xc_n = xconv[off];
    float z_n  = xz[zoff];
    float Bn[16], Cn[16];
    {
        const float* row = BC + baseBL * 32;
        #pragma unroll
        for (int i = 0; i < 16; ++i) { Bn[i] = row[i]; Cn[i] = row[16 + i]; }
    }

    for (int t = 0; t < Ll; ++t) {
        float dtv = dt_n + bias;
        const float xc = xc_n;
        const float zz = z_n;
        float Bf[16], Cf[16];
        #pragma unroll
        for (int i = 0; i < 16; ++i) { Bf[i] = Bn[i]; Cf[i] = Cn[i]; }

        if (t + 1 < Ll) {                       // prefetch t+1 into registers
            dt_n = dtlin[off + Dd];
            xc_n = xconv[off + Dd];
            z_n  = xz[zoff + 2L * Dd];
            const float* row = BC + (baseBL + t + 1) * 32;
            #pragma unroll
            for (int i = 0; i < 16; ++i) { Bn[i] = row[i]; Cn[i] = row[16 + i]; }
        }

        dtv = (dtv > 20.0f) ? dtv : __logf(1.0f + __expf(dtv));   // softplus
        float y = Dp * xc;
        const float dxc = dtv * xc;
        #pragma unroll
        for (int s = 0; s < 16; ++s) {
            float dA = __expf(dtv * Amat[s]);
            h[s] = dA * h[s] + Bf[s] * dxc;
            y += h[s] * Cf[s];
        }
        g[off] = y * silu_f(zz);
        off  += Dd;
        zoff += 2L * Dd;
    }
}

extern "C" void kernel_launch(void* const* d_in, const int* in_sizes, int n_in,
                              void* d_out, int out_size, void* d_ws, size_t ws_size,
                              hipStream_t stream) {
    const float* x       = (const float*)d_in[0];
    const float* w_in    = (const float*)d_in[1];
    const float* conv_w  = (const float*)d_in[2];
    const float* conv_b  = (const float*)d_in[3];
    const float* w_x     = (const float*)d_in[4];
    const float* w_dt    = (const float*)d_in[5];
    const float* b_dt    = (const float*)d_in[6];
    const float* A_log   = (const float*)d_in[7];
    const float* D_param = (const float*)d_in[8];
    const float* w_out   = (const float*)d_in[9];
    float* out = (float*)d_out;

    const int Bb = 4, Ll = 2048, Dd = 768, Ss = 16;
    const int Mr = Bb * Ll;  // 8192 rows

    float* ws    = (float*)d_ws;
    float* xz    = ws;                                 // Mr * 2D
    float* xconv = xz    + (size_t)Mr * 2 * Dd;        // Mr * D
    float* dtlin = xconv + (size_t)Mr * Dd;            // Mr * D
    float* BCb   = dtlin + (size_t)Mr * Dd;            // Mr * 2S
    float* gbuf  = BCb   + (size_t)Mr * 2 * Ss;        // Mr * D

    dim3 blk(256);

    // 1) xz = x @ w_in   (8192 x 768 x 1536)
    {
        dim3 grid((2 * Dd) / TILE_N, Mr / TILE_M);
        gemm_bf16_wmma<false><<<grid, blk, 0, stream>>>(x, w_in, xz, Mr, 2 * Dd, Dd);
    }
    // 2) x_conv = silu(depthwise_conv(x_gate) + b)
    {
        long total4 = (long)Mr * Dd / 4;
        conv_silu_kernel<<<(unsigned)((total4 + 255) / 256), blk, 0, stream>>>(
            xz, conv_w, conv_b, xconv, Ll, Dd, total4);
    }
    // 3) dtlin = x_conv @ w_dt   (8192 x 768 x 768)
    {
        dim3 grid(Dd / TILE_N, Mr / TILE_M);
        gemm_bf16_wmma<false><<<grid, blk, 0, stream>>>(xconv, w_dt, dtlin, Mr, Dd, Dd);
    }
    // 4) BC = x_conv @ w_x       (8192 x 768 x 32), N=32 -> edge path
    {
        dim3 grid(1, Mr / TILE_M);
        gemm_bf16_wmma<true><<<grid, blk, 0, stream>>>(xconv, w_x, BCb, Mr, 2 * Ss, Dd);
    }
    // 5) selective scan + gating -> gbuf
    {
        dim3 grid(Bb, Dd / 64);
        scan_kernel<<<grid, dim3(64), 0, stream>>>(dtlin, b_dt, BCb, xconv,
                                                   A_log, D_param, xz, gbuf, Ll, Dd);
    }
    // 6) out = gbuf @ w_out      (8192 x 768 x 768)
    {
        dim3 grid(Dd / TILE_N, Mr / TILE_M);
        gemm_bf16_wmma<false><<<grid, blk, 0, stream>>>(gbuf, w_out, out, Mr, Dd, Dd);
    }
}
